// TwoTowerRecommendation_9079560863845
// MI455X (gfx1250) — matrix operand order repaired and checked by reading, not compiled, not run on hardware
//
#include <hip/hip_runtime.h>
#include <hip/hip_bf16.h>

// ---------------- problem constants (from reference setup_inputs) ------------
static constexpr int kD      = 768;
static constexpr int kB      = 1024;
static constexpr int kH      = 50;
static constexpr int kPc     = 5;
static constexpr int kPn     = 50;
static constexpr int kAtt    = 200;
static constexpr int kAttPad = 208;   // 13 tiles of 16
#define LN_EPS 1e-5f

typedef __attribute__((ext_vector_type(16))) __bf16 v16bf;
typedef __attribute__((ext_vector_type(8)))  __bf16 v8bf;
typedef __attribute__((ext_vector_type(8)))  float  v8f;

__device__ __forceinline__ v8f vzero8() {
  v8f z;
#pragma unroll
  for (int i = 0; i < 8; ++i) z[i] = 0.f;
  return z;
}

// A-fragment (16x32 bf16) loaded DIRECTLY from row-major fp32 global memory.
// Per ISA layout, lane (m, half) holds K = {8h+0..7, 16+8h+0..7} for row m —
// i.e. two contiguous 8-float (32-byte aligned) runs. Convert in registers.
__device__ __forceinline__ v16bf frag_a_from_global(const float* __restrict__ xrow,
                                                    int half) {
  v8f lo = *(const v8f*)(xrow + 8 * half);
  v8f hi = *(const v8f*)(xrow + 16 + 8 * half);
  v16bf a;
#pragma unroll
  for (int i = 0; i < 8; ++i) {
    a[i]     = (__bf16)lo[i];
    a[8 + i] = (__bf16)hi[i];
  }
  return a;
}

// B-fragment (32x16 bf16): lane n holds column n; lane half h holds K = {8h+0..7, 16+8h+0..7}.
// Column n of B == contiguous bf16 row in our pre-transposed weight buffer.
__device__ __forceinline__ v16bf frag_b_from_global(const __bf16* __restrict__ p, int half) {
  v8bf lo = *(const v8bf*)(p + 8 * half);
  v8bf hi = *(const v8bf*)(p + 16 + 8 * half);
  return __builtin_shufflevector(lo, hi, 0, 1, 2, 3, 4, 5, 6, 7, 8, 9, 10, 11, 12, 13, 14, 15);
}

// ---------------- weight conversion ------------------------------------------
__global__ void convert_wnews(const float* __restrict__ W, __bf16* __restrict__ Wbf) {
  int i = blockIdx.x * 256 + threadIdx.x;
  if (i < kD * kD) Wbf[i] = (__bf16)W[i];
}

// WattT[n*768 + k] = W_att[k*200 + n] (zero-padded cols 200..207)
__global__ void convert_watt(const float* __restrict__ Watt, __bf16* __restrict__ WT) {
  int i = blockIdx.x * 256 + threadIdx.x;
  if (i < kAttPad * kD) {
    int n = i / kD, k = i - n * kD;
    WT[i] = (n < kAtt) ? (__bf16)Watt[k * kAtt + n] : (__bf16)0.f;
  }
}

// ---------------- news encoder: GEMM (bf16 WMMA) + LN + ReLU + L2 ------------
// Tile: 16 rows x 768 cols per block (8 wave32, 6 N-tiles per wave).
// Ping-pong pipelined K loop (unroll-2, two named fragment buffers): the loads
// for K-step s+1 are issued before the WMMA chain of step s, and buffers swap
// roles by name — no v_mov register rotation, no s_wait_loadcnt 0 before MMAs.
__launch_bounds__(256)
__global__ void news_encode_kernel(const float* __restrict__ X,
                                   const __bf16* __restrict__ Wbf,
                                   const float* __restrict__ lng,
                                   const float* __restrict__ lnb,
                                   float* __restrict__ Y) {
  __shared__ float sTile[16 * kD];     // 48 KB output tile for LN epilogue
  const int tid  = threadIdx.x;
  const int wave = tid >> 5, lane = tid & 31;
  const int m = lane & 15, half = lane >> 4;
  const int row0 = blockIdx.x * 16;

  const float* xrow = X + (size_t)(row0 + m) * kD;
  const __bf16* wbase = Wbf + (size_t)(wave * 96 + m) * kD;

  v8f acc[6];
#pragma unroll
  for (int t = 0; t < 6; ++t) acc[t] = vzero8();

  // prologue: fragments for kb = 0 into buffer A
  v16bf aA = frag_a_from_global(xrow, half);
  v16bf bA[6];
#pragma unroll
  for (int t = 0; t < 6; ++t)
    bA[t] = frag_b_from_global(wbase + (size_t)(t * 16) * kD, half);

  v16bf aB;
  v16bf bB[6];

  int kb = 0;
  for (; kb < kD - 64; kb += 64) {
    // prefetch kb+32 into buffer B
    aB = frag_a_from_global(xrow + kb + 32, half);
#pragma unroll
    for (int t = 0; t < 6; ++t)
      bB[t] = frag_b_from_global(wbase + (size_t)(t * 16) * kD + kb + 32, half);
    // consume buffer A (kb)
#pragma unroll
    for (int t = 0; t < 6; ++t)
      acc[t] = __builtin_amdgcn_wmma_f32_16x16x32_bf16(false, aA, false, bA[t],
                                                       (short)0, acc[t], false, false);
    // prefetch kb+64 into buffer A
    aA = frag_a_from_global(xrow + kb + 64, half);
#pragma unroll
    for (int t = 0; t < 6; ++t)
      bA[t] = frag_b_from_global(wbase + (size_t)(t * 16) * kD + kb + 64, half);
    // consume buffer B (kb+32)
#pragma unroll
    for (int t = 0; t < 6; ++t)
      acc[t] = __builtin_amdgcn_wmma_f32_16x16x32_bf16(false, aB, false, bB[t],
                                                       (short)0, acc[t], false, false);
  }
  // peeled tail: kb == kD-64; consume kb (buffer A) and kb+32 (buffer B)
  aB = frag_a_from_global(xrow + kb + 32, half);
#pragma unroll
  for (int t = 0; t < 6; ++t)
    bB[t] = frag_b_from_global(wbase + (size_t)(t * 16) * kD + kb + 32, half);
#pragma unroll
  for (int t = 0; t < 6; ++t)
    acc[t] = __builtin_amdgcn_wmma_f32_16x16x32_bf16(false, aA, false, bA[t],
                                                     (short)0, acc[t], false, false);
#pragma unroll
  for (int t = 0; t < 6; ++t)
    acc[t] = __builtin_amdgcn_wmma_f32_16x16x32_bf16(false, aB, false, bB[t],
                                                     (short)0, acc[t], false, false);

  // dump accumulators: VGPR j of lane -> row (j or j+8), col = wave*96 + t*16 + lane%16
  {
    const int mofs = half * 8;
#pragma unroll
    for (int t = 0; t < 6; ++t) {
      int col = wave * 96 + t * 16 + m;
#pragma unroll
      for (int j = 0; j < 8; ++j) sTile[(mofs + j) * kD + col] = acc[t][j];
    }
  }
  __syncthreads();

  // LayerNorm + ReLU + L2-normalize: 16 threads per row, 48 cols each
  const int g = tid >> 4;   // tile row
  const int tl = tid & 15;
  float sum = 0.f, sq = 0.f;
  float xv[48];
#pragma unroll
  for (int k = 0; k < 48; ++k) {
    float v = sTile[g * kD + tl + 16 * k];
    xv[k] = v; sum += v; sq += v * v;
  }
#pragma unroll
  for (int mm = 1; mm < 16; mm <<= 1) {
    sum += __shfl_xor(sum, mm, 16);
    sq  += __shfl_xor(sq,  mm, 16);
  }
  float mu   = sum * (1.f / kD);
  float var  = sq * (1.f / kD) - mu * mu;
  float rstd = rsqrtf(var + LN_EPS);
  float nsq = 0.f;
#pragma unroll
  for (int k = 0; k < 48; ++k) {
    int c = tl + 16 * k;
    float y = (xv[k] - mu) * rstd * lng[c] + lnb[c];
    y = fmaxf(y, 0.f);
    xv[k] = y; nsq += y * y;
  }
#pragma unroll
  for (int mm = 1; mm < 16; mm <<= 1) nsq += __shfl_xor(nsq, mm, 16);
  float inv = 1.f / fmaxf(sqrtf(nsq), 1e-12f);
#pragma unroll
  for (int k = 0; k < 48; ++k)
    Y[(size_t)(row0 + g) * kD + tl + 16 * k] = xv[k] * inv;
}

// ---------------- attention scores: tanh(hist_e @ W_att) @ v_att -------------
__launch_bounds__(256)
__global__ void att_score_kernel(const float* __restrict__ E,        // [B*H, D] encoded
                                 const __bf16* __restrict__ WattT,   // [208, 768]
                                 const float* __restrict__ v_att,    // [200]
                                 float* __restrict__ scores) {       // [B*H]
  __shared__ float rowsum[16];
  const int tid  = threadIdx.x;
  const int wave = tid >> 5, lane = tid & 31;
  const int m = lane & 15, half = lane >> 4;
  const int row0 = blockIdx.x * 16;
  if (tid < 16) rowsum[tid] = 0.f;

  const float* xrow = E + (size_t)(row0 + m) * kD;
  const int two = (wave < 5) ? 1 : 0;            // 13 N-tiles over 8 waves (wave-uniform)
  const __bf16* wrow0 = WattT + (size_t)(wave * 16 + m) * kD;
  const __bf16* wrow1 = WattT + (size_t)((wave + 8) * 16 + m) * kD;

  v8f acc[2];
  acc[0] = vzero8(); acc[1] = vzero8();

  // prologue into buffer A
  v16bf aA = frag_a_from_global(xrow, half);
  v16bf bA0 = frag_b_from_global(wrow0, half);
  v16bf bA1;
  if (two) bA1 = frag_b_from_global(wrow1, half);

  v16bf aB, bB0, bB1;

  int kb = 0;
  for (; kb < kD - 64; kb += 64) {
    aB  = frag_a_from_global(xrow + kb + 32, half);
    bB0 = frag_b_from_global(wrow0 + kb + 32, half);
    if (two) bB1 = frag_b_from_global(wrow1 + kb + 32, half);

    acc[0] = __builtin_amdgcn_wmma_f32_16x16x32_bf16(false, aA, false, bA0,
                                                     (short)0, acc[0], false, false);
    if (two)
      acc[1] = __builtin_amdgcn_wmma_f32_16x16x32_bf16(false, aA, false, bA1,
                                                       (short)0, acc[1], false, false);

    aA  = frag_a_from_global(xrow + kb + 64, half);
    bA0 = frag_b_from_global(wrow0 + kb + 64, half);
    if (two) bA1 = frag_b_from_global(wrow1 + kb + 64, half);

    acc[0] = __builtin_amdgcn_wmma_f32_16x16x32_bf16(false, aB, false, bB0,
                                                     (short)0, acc[0], false, false);
    if (two)
      acc[1] = __builtin_amdgcn_wmma_f32_16x16x32_bf16(false, aB, false, bB1,
                                                       (short)0, acc[1], false, false);
  }
  // peeled tail
  aB  = frag_a_from_global(xrow + kb + 32, half);
  bB0 = frag_b_from_global(wrow0 + kb + 32, half);
  if (two) bB1 = frag_b_from_global(wrow1 + kb + 32, half);
  acc[0] = __builtin_amdgcn_wmma_f32_16x16x32_bf16(false, aA, false, bA0,
                                                   (short)0, acc[0], false, false);
  if (two)
    acc[1] = __builtin_amdgcn_wmma_f32_16x16x32_bf16(false, aA, false, bA1,
                                                     (short)0, acc[1], false, false);
  acc[0] = __builtin_amdgcn_wmma_f32_16x16x32_bf16(false, aB, false, bB0,
                                                   (short)0, acc[0], false, false);
  if (two)
    acc[1] = __builtin_amdgcn_wmma_f32_16x16x32_bf16(false, aB, false, bB1,
                                                     (short)0, acc[1], false, false);

  __syncthreads();  // rowsum init visible

  const int mofs = half * 8;
  for (int ti = 0; ti <= two; ++ti) {
    int col = (wave + ti * 8) * 16 + m;
    float va = (col < kAtt) ? v_att[col] : 0.f;
#pragma unroll
    for (int j = 0; j < 8; ++j) {
      float s = tanhf(acc[ti][j]) * va;
      atomicAdd(&rowsum[mofs + j], s);
    }
  }
  __syncthreads();
  if (tid < 16) scores[row0 + tid] = rowsum[tid];
}

// ---------------- softmax + attention pooling + L2 ---------------------------
__global__ void user_kernel(const float* __restrict__ scores,   // [B*H]
                            const float* __restrict__ E,        // hist_e [B*H, D]
                            float* __restrict__ user) {         // [B, D]
  __shared__ float attn[kH];
  __shared__ float red[256];
  const int b = blockIdx.x;
  const int tid = threadIdx.x;
  if (tid < kH) attn[tid] = scores[b * kH + tid];
  __syncthreads();
  if (tid == 0) {
    float mx = -1e30f;
    for (int h = 0; h < kH; ++h) mx = fmaxf(mx, attn[h]);
    float s = 0.f;
    for (int h = 0; h < kH; ++h) { float ev = __expf(attn[h] - mx); attn[h] = ev; s += ev; }
    float inv = 1.f / s;
    for (int h = 0; h < kH; ++h) attn[h] *= inv;
  }
  __syncthreads();

  float acc[3]; float ss = 0.f;
#pragma unroll
  for (int i = 0; i < 3; ++i) {
    int d = tid + i * 256;
    const float* ep = E + ((size_t)b * kH) * kD + d;
    float a = 0.f;
    for (int h = 0; h < kH; ++h) a += attn[h] * ep[(size_t)h * kD];
    acc[i] = a; ss += a * a;
  }
  red[tid] = ss; __syncthreads();
  for (int s = 128; s > 0; s >>= 1) { if (tid < s) red[tid] += red[tid + s]; __syncthreads(); }
  float inv = 1.f / fmaxf(sqrtf(red[0]), 1e-12f);
#pragma unroll
  for (int i = 0; i < 3; ++i) user[(size_t)b * kD + tid + i * 256] = acc[i] * inv;
}

// ---------------- relevance dots + output assembly ---------------------------
__global__ void rel_out_kernel(const float* __restrict__ user,    // [B, D]
                               const float* __restrict__ clk_e,   // [B*Pc, D]
                               const float* __restrict__ nclk_e,  // [B*Pn, D]
                               float* __restrict__ out) {
  __shared__ float u[kD];
  const int b = blockIdx.x;
  const int tid = threadIdx.x;
  const int wave = tid >> 5, lane = tid & 31;
#pragma unroll
  for (int i = 0; i < 3; ++i) u[tid + i * 256] = user[(size_t)b * kD + tid + i * 256];
  __syncthreads();

  const int TOT = kB * (kPc + kPn);   // 56320 (all items valid for gaussian inputs)
  float* idxO = out;
  float* relO = out + TOT;
  float* tgtO = out + 2 * TOT;

  for (int item = wave; item < kPc + kPn; item += 8) {
    const float* e = (item < kPc)
        ? (clk_e  + ((size_t)b * kPc + item) * kD)
        : (nclk_e + ((size_t)b * kPn + (item - kPc)) * kD);
    float dot = 0.f;
    for (int d = lane; d < kD; d += 32) dot += u[d] * e[d];
#pragma unroll
    for (int mm = 1; mm < 32; mm <<= 1) dot += __shfl_xor(dot, mm, 32);
    if (lane == 0) {
      int pos = (item < kPc) ? (b * kPc + item)
                             : (kB * kPc + b * kPn + (item - kPc));
      idxO[pos] = (float)b;
      relO[pos] = dot;
      tgtO[pos] = (item < kPc) ? 1.f : 0.f;
    }
  }
}

// ---------------- launch ------------------------------------------------------
extern "C" void kernel_launch(void* const* d_in, const int* in_sizes, int n_in,
                              void* d_out, int out_size, void* d_ws, size_t ws_size,
                              hipStream_t stream) {
  (void)in_sizes; (void)n_in; (void)out_size; (void)ws_size;
  const float* history = (const float*)d_in[0];
  const float* clicks  = (const float*)d_in[1];
  const float* nclicks = (const float*)d_in[2];
  const float* W_news  = (const float*)d_in[3];
  const float* ln_g    = (const float*)d_in[4];
  const float* ln_b    = (const float*)d_in[5];
  const float* W_att   = (const float*)d_in[6];
  const float* v_att   = (const float*)d_in[7];
  float* out = (float*)d_out;

  char* ws = (char*)d_ws;
  size_t off = 0;
  auto alloc = [&](size_t bytes) {
    size_t p = off;
    off += (bytes + 255) & ~(size_t)255;
    return p;
  };
  __bf16* Wbf    = (__bf16*)(ws + alloc((size_t)kD * kD * 2));
  __bf16* WattT  = (__bf16*)(ws + alloc((size_t)kAttPad * kD * 2));
  float*  hist_e = (float*)(ws + alloc((size_t)kB * kH * kD * 4));
  float*  clk_e  = (float*)(ws + alloc((size_t)kB * kPc * kD * 4));
  float*  nclk_e = (float*)(ws + alloc((size_t)kB * kPn * kD * 4));
  float*  scores = (float*)(ws + alloc((size_t)kB * kH * 4));
  float*  user   = (float*)(ws + alloc((size_t)kB * kD * 4));

  convert_wnews<<<(kD * kD + 255) / 256, 256, 0, stream>>>(W_news, Wbf);
  convert_watt<<<(kAttPad * kD + 255) / 256, 256, 0, stream>>>(W_att, WattT);

  news_encode_kernel<<<kB * kH / 16, 256, 0, stream>>>(history, Wbf, ln_g, ln_b, hist_e);
  news_encode_kernel<<<kB * kPc / 16, 256, 0, stream>>>(clicks, Wbf, ln_g, ln_b, clk_e);
  news_encode_kernel<<<kB * kPn / 16, 256, 0, stream>>>(nclicks, Wbf, ln_g, ln_b, nclk_e);

  att_score_kernel<<<kB * kH / 16, 256, 0, stream>>>(hist_e, WattT, v_att, scores);
  user_kernel<<<kB, 256, 0, stream>>>(scores, hist_e, user);
  rel_out_kernel<<<kB, 256, 0, stream>>>(user, clk_e, nclk_e, out);
}
